// TextGatedMambaBlock_46772193853540
// MI455X (gfx1250) — matrix operand order repaired and checked by reading, not compiled, not run on hardware
//
#include <hip/hip_runtime.h>
#include <hip/hip_bf16.h>

// ---------------- Problem constants ----------------
#define BATCH   8
#define CCH     768            // C
#define LSEQ    1024           // H*W
#define DIN     1536           // D_INNER
#define NXZ     3072           // 2*D_INNER
#define DTR     48             // DT_RANK
#define DTRP    64             // DT_RANK padded to K multiple of 32
#define DST     16             // D_STATE
#define MROWS   (BATCH*LSEQ)   // 8192

typedef unsigned int  u32;
typedef unsigned short u16;
typedef __attribute__((ext_vector_type(4)))  u32     u32x4;
typedef __attribute__((ext_vector_type(4)))  int     i32x4;
typedef __attribute__((ext_vector_type(8)))  int     i32x8;
typedef __attribute__((ext_vector_type(16))) __bf16  v16bf;
typedef __attribute__((ext_vector_type(8)))  float   v8f;

#if __has_builtin(__builtin_amdgcn_tensor_load_to_lds) && \
    __has_builtin(__builtin_amdgcn_s_wait_tensorcnt)
#define USE_TDM 1
#else
#define USE_TDM 0
#endif

// ---------------- helpers ----------------
__device__ __forceinline__ u16 f2bf(float f) {
    u32 u = __float_as_uint(f);
    u32 r = u + 0x7FFFu + ((u >> 16) & 1u);   // RNE
    return (u16)(r >> 16);
}
__device__ __forceinline__ float bf2f(u16 h) {
    return __uint_as_float(((u32)h) << 16);
}
__device__ __forceinline__ float sigmoidf_(float x) {
    return 1.0f / (1.0f + __expf(-x));
}
__device__ __forceinline__ u32 rfl(u32 x) {
    return (u32)__builtin_amdgcn_readfirstlane((int)x);
}

#if USE_TDM
// Issue one TDM 2-D tile load: tile_dim0=32 elems (64B), tile_dim1=16 rows,
// bf16 (data_size=2B), LDS pad 16B every 64B -> 80B LDS row stride.
__device__ __forceinline__ void tdm_load_tile(
    u32 lds_byte_off, const u16* gsrc, int rows_valid, int k_valid, int ld_elems)
{
    unsigned long long ga = (unsigned long long)gsrc;
    u32 ga_lo = rfl((u32)ga);
    u32 ga_hi = rfl((u32)(ga >> 32));
    u32 ldso  = rfl(lds_byte_off);
    u32 td0   = rfl((u32)k_valid);                       // tensor_dim0 (elems)
    u32 td1   = rfl((u32)(rows_valid < 0 ? 0 : rows_valid)); // tensor_dim1
    u32 st0   = (u32)ld_elems;                           // dim0 stride (elems)

    u32x4 g0;
    g0[0] = 1u;                                          // count=1
    g0[1] = ldso;                                        // lds_addr
    g0[2] = ga_lo;                                       // global_addr[31:0]
    g0[3] = (ga_hi & 0x01FFFFFFu) | (2u << 30);          // ga[56:32] | type=2

    i32x8 g1;
    g1[0] = (int)((1u << 16)     // data_size = 2 bytes
                | (1u << 20)     // pad_enable
                | (3u << 22)     // pad_interval: 16 DWORDs (64B)
                | (3u << 25));   // pad_amount:   4 DWORDs (16B)
    g1[1] = (int)((td0 & 0xFFFFu) << 16);                     // tensor_dim0 lo16
    g1[2] = (int)((td0 >> 16) | ((td1 & 0xFFFFu) << 16));     // dim0 hi | dim1 lo
    g1[3] = (int)((td1 >> 16) | (32u << 16));                 // dim1 hi | tile_dim0=32
    g1[4] = (int)16u;                                         // tile_dim1=16, tile_dim2=0
    g1[5] = (int)st0;                                         // dim0_stride lo32
    g1[6] = 0;                                                // dim0_stride hi | dim1_stride lo
    g1[7] = 0;

    i32x4 gz = {0, 0, 0, 0};
#if defined(__clang_major__) && (__clang_major__ >= 23)
    i32x8 gz8 = {0, 0, 0, 0, 0, 0, 0, 0};
    __builtin_amdgcn_tensor_load_to_lds(g0, g1, gz, gz, gz8, 0);
#else
    __builtin_amdgcn_tensor_load_to_lds(g0, g1, gz, gz, 0);
#endif
}
#endif

// ---------------- tiny kernels ----------------

__global__ __launch_bounds__(256) void text_proj_kernel(
    const float* __restrict__ text, const float* __restrict__ Wt,
    const float* __restrict__ bt, float* __restrict__ t)
{
    int idx = blockIdx.x * 256 + threadIdx.x;
    if (idx >= BATCH * CCH) return;
    int b = idx / CCH, c = idx % CCH;
    const float* te = text + (size_t)b * CCH;
    const float* w  = Wt   + (size_t)c * CCH;
    float acc = bt[c];
    for (int k = 0; k < CCH; ++k) acc += te[k] * w[k];
    t[idx] = acc;
}

__global__ __launch_bounds__(256) void f32_to_bf16_kernel(
    const float* __restrict__ src, u16* __restrict__ dst, int n)
{
    int i = blockIdx.x * 256 + threadIdx.x;
    if (i < n) dst[i] = f2bf(src[i]);
}

__global__ __launch_bounds__(256) void wdt_pad_kernel(
    const float* __restrict__ W, u16* __restrict__ dst)
{
    int i = blockIdx.x * 256 + threadIdx.x;
    if (i >= DIN * DTRP) return;
    int n = i >> 6, k = i & 63;
    dst[i] = (k < DTR) ? f2bf(W[(size_t)n * DTR + k]) : (u16)0;
}

__global__ __launch_bounds__(256) void pack_dt_kernel(
    const float* __restrict__ dbl, u16* __restrict__ dst)
{
    int i = blockIdx.x * 256 + threadIdx.x;
    if (i >= MROWS * DTRP) return;
    int m = i >> 6, k = i & 63;
    dst[i] = (k < DTR) ? f2bf(dbl[(size_t)m * 80 + k]) : (u16)0;
}

__global__ __launch_bounds__(256) void aexp_kernel(
    const float* __restrict__ A_log, float* __restrict__ Aexp)
{
    int i = blockIdx.x * 256 + threadIdx.x;
    if (i < DIN * DST) Aexp[i] = -__expf(A_log[i]);
}

// gate + layernorm, emit xn bf16 (8192 x 768). One block per token (b,l).
__global__ __launch_bounds__(256) void gate_ln_kernel(
    const float* __restrict__ vis, const float* __restrict__ t,
    const float* __restrict__ ln_g, const float* __restrict__ ln_b,
    u16* __restrict__ xn)
{
    int m = blockIdx.x;
    int b = m >> 10, l = m & (LSEQ - 1);
    __shared__ float vm[CCH];
    __shared__ float red[16];
    __shared__ float stats[2];

    float s = 0.f, s2 = 0.f;
    for (int c = threadIdx.x; c < CCH; c += 256) {
        float v  = vis[(((size_t)b * CCH + c) << 10) + l];
        float tv = t[b * CCH + c];
        float x  = v * sigmoidf_(v * tv);
        vm[c] = x; s += x; s2 += x * x;
    }
    for (int off = 16; off > 0; off >>= 1) {
        s  += __shfl_down(s,  off, 32);
        s2 += __shfl_down(s2, off, 32);
    }
    int wv = threadIdx.x >> 5, ln = threadIdx.x & 31;
    if (ln == 0) { red[wv] = s; red[8 + wv] = s2; }
    __syncthreads();
    if (threadIdx.x == 0) {
        float ts = 0.f, ts2 = 0.f;
        for (int w = 0; w < 8; ++w) { ts += red[w]; ts2 += red[8 + w]; }
        float mu  = ts  * (1.0f / CCH);
        float var = ts2 * (1.0f / CCH) - mu * mu;
        stats[0] = mu; stats[1] = rsqrtf(var + 1e-5f);
    }
    __syncthreads();
    float mu = stats[0], rs = stats[1];
    for (int c = threadIdx.x; c < CCH; c += 256) {
        float x = (vm[c] - mu) * rs * ln_g[c] + ln_b[c];
        xn[(size_t)m * CCH + c] = f2bf(x);
    }
}

// depthwise causal conv (k=4) + SiLU
__global__ __launch_bounds__(256) void conv_silu_kernel(
    const u16* __restrict__ xz, const float* __restrict__ Wc,
    const float* __restrict__ bc, u16* __restrict__ xc)
{
    int idx = blockIdx.x * 256 + threadIdx.x;
    if (idx >= MROWS * DIN) return;
    int d = idx % DIN, m = idx / DIN, l = m & (LSEQ - 1);
    float acc = bc[d];
    #pragma unroll
    for (int k = 0; k < 4; ++k) {
        int ll = l + k - 3;
        if (ll >= 0)
            acc += bf2f(xz[(size_t)(m + k - 3) * NXZ + d]) * Wc[d * 4 + k];
    }
    float y = acc * sigmoidf_(acc);
    xc[idx] = f2bf(y);
}

// selective scan: one lane per (b,d); fused (ys + xc*D)*silu(z) -> y bf16
__global__ __launch_bounds__(256) void scan_kernel(
    const float* __restrict__ dt, const u16* __restrict__ xc,
    const float* __restrict__ dbl, const float* __restrict__ Aexp,
    const float* __restrict__ Dp, const u16* __restrict__ xz,
    u16* __restrict__ ybf)
{
    int idx = blockIdx.x * 256 + threadIdx.x;       // b*1536 + d
    int b = idx / DIN, d = idx % DIN;
    float Ae[DST], h[DST];
    #pragma unroll
    for (int s = 0; s < DST; ++s) { Ae[s] = Aexp[d * DST + s]; h[s] = 0.f; }
    float Dd = Dp[d];
    for (int l = 0; l < LSEQ; ++l) {
        int m = (b << 10) + l;
        float dtt = dt[(size_t)m * DIN + d];
        float xt  = bf2f(xc[(size_t)m * DIN + d]);
        const float* bcv = dbl + (size_t)m * 80 + DTR;   // B [48..63], C [64..79]
        float y = 0.f;
        float dx = dtt * xt;
        #pragma unroll
        for (int s = 0; s < DST; ++s) {
            float dA = __expf(dtt * Ae[s]);
            h[s] = h[s] * dA + dx * bcv[s];
            y += h[s] * bcv[DST + s];
        }
        float z  = bf2f(xz[(size_t)m * NXZ + DIN + d]);
        float yv = (y + xt * Dd) * (z * sigmoidf_(z));
        ybf[(size_t)m * DIN + d] = f2bf(yv);
    }
}

// ---------------- WMMA bf16 GEMM (TDM double-buffered) ----------------
// C[m,n] = sum_k A[m,k]*Bt[n,k]; A (MxK), Bt (NxK) bf16 row-major.
// Block 256 thr = 8 waves; tile 128x128; K-step 32; TDM loads tiles to LDS
// with hardware row padding (64B data + 16B pad = 80B = LDSW*2).
// MODE 0: f32; 1: bf16; 2: softplus(acc+bias[n]) f32; 3: transpose+residual.
#define LDSW 40

template<int MODE>
__global__ __launch_bounds__(256) void gemm_bf16_kernel(
    const u16* __restrict__ A, const u16* __restrict__ Bt,
    void* __restrict__ Cout, const float* __restrict__ bias,
    const float* __restrict__ resid,
    int M, int N, int K, int lda, int ldb, int ldc)
{
    __shared__ __align__(16) u16 sA[2][128 * LDSW];
    __shared__ __align__(16) u16 sB[2][128 * LDSW];

    const int tid  = threadIdx.x;
    const int wv   = tid >> 5, lane = tid & 31;
    const int hh   = lane >> 4, lrow = lane & 15;
    const int wm   = wv >> 2,  wn   = wv & 3;
    const int mBlk = blockIdx.x * 128, nBlk = blockIdx.y * 128;

    const v8f vzero = {0.f, 0.f, 0.f, 0.f, 0.f, 0.f, 0.f, 0.f};
    v8f acc[4][2];
    #pragma unroll
    for (int i = 0; i < 4; ++i)
        #pragma unroll
        for (int j = 0; j < 2; ++j) acc[i][j] = vzero;

    union BF { v16bf v; u32x4 q[2]; };

#if USE_TDM
    const int subRow = wv * 16;      // each wave DMAs 16 rows of each tile
    const u32 ldsA0 = (u32)(uintptr_t)&sA[0][0];
    const u32 ldsB0 = (u32)(uintptr_t)&sB[0][0];
    auto issue_tiles = [&](int buf, int k0) {
        tdm_load_tile(ldsA0 + (u32)(buf * 128 + subRow) * (LDSW * 2),
                      A + (size_t)(mBlk + subRow) * lda + k0,
                      M - (mBlk + subRow), K - k0, lda);
        tdm_load_tile(ldsB0 + (u32)(buf * 128 + subRow) * (LDSW * 2),
                      Bt + (size_t)(nBlk + subRow) * ldb + k0,
                      N - (nBlk + subRow), K - k0, ldb);
    };
    issue_tiles(0, 0);
#else
    const int r    = tid >> 1;
    const int coff = (tid & 1) * 16;
    auto issue_tiles = [&](int buf, int k0) {
        const u16* ag = A + (size_t)(mBlk + r) * lda + k0 + coff;
        u32x4 a0 = *(const u32x4*)ag;
        u32x4 a1 = *(const u32x4*)(ag + 8);
        *(u32x4*)&sA[buf][r * LDSW + coff]     = a0;
        *(u32x4*)&sA[buf][r * LDSW + coff + 8] = a1;
        u32x4 b0 = {0, 0, 0, 0}, b1 = {0, 0, 0, 0};
        int nrow = nBlk + r;
        if (nrow < N) {
            const u16* bg = Bt + (size_t)nrow * ldb + k0 + coff;
            b0 = *(const u32x4*)bg;
            b1 = *(const u32x4*)(bg + 8);
        }
        *(u32x4*)&sB[buf][r * LDSW + coff]     = b0;
        *(u32x4*)&sB[buf][r * LDSW + coff + 8] = b1;
    };
    issue_tiles(0, 0);
#endif

    int buf = 0;
    for (int k0 = 0; k0 < K; k0 += 32) {
#if USE_TDM
        __builtin_amdgcn_s_wait_tensorcnt(0);   // own wave's tile slices landed
#endif
        __syncthreads();                        // all waves' slices visible
        if (k0 + 32 < K) issue_tiles(buf ^ 1, k0 + 32);

        const u16* aL = &sA[buf][0];
        const u16* bL = &sB[buf][0];
        BF bfr[2];
        #pragma unroll
        for (int j = 0; j < 2; ++j) {
            int n = wn * 32 + j * 16 + lrow;
            bfr[j].q[0] = *(const u32x4*)&bL[n * LDSW + hh * 16];
            bfr[j].q[1] = *(const u32x4*)&bL[n * LDSW + hh * 16 + 8];
        }
        #pragma unroll
        for (int i = 0; i < 4; ++i) {
            BF af;
            int row = wm * 64 + i * 16 + lrow;
            af.q[0] = *(const u32x4*)&aL[row * LDSW + hh * 16];
            af.q[1] = *(const u32x4*)&aL[row * LDSW + hh * 16 + 8];
            #pragma unroll
            for (int j = 0; j < 2; ++j) {
                acc[i][j] = __builtin_amdgcn_wmma_f32_16x16x32_bf16(
                    false, af.v, false, bfr[j].v, (short)0, acc[i][j],
                    false, false);
            }
        }
        buf ^= 1;
    }

    // epilogue: element v of v8f -> m = base + hh*8 + v, n = base + lrow
    #pragma unroll
    for (int i = 0; i < 4; ++i) {
        int mB = mBlk + wm * 64 + i * 16 + hh * 8;
        #pragma unroll
        for (int j = 0; j < 2; ++j) {
            int n = nBlk + wn * 32 + j * 16 + lrow;
            if (n < N) {
                #pragma unroll
                for (int v = 0; v < 8; ++v) {
                    int m = mB + v;
                    float val = acc[i][j][v];
                    if constexpr (MODE == 0) {
                        ((float*)Cout)[(size_t)m * ldc + n] = val;
                    } else if constexpr (MODE == 1) {
                        ((u16*)Cout)[(size_t)m * ldc + n] = f2bf(val);
                    } else if constexpr (MODE == 2) {
                        float x = val + bias[n];
                        ((float*)Cout)[(size_t)m * ldc + n] =
                            (x > 20.f) ? x : log1pf(__expf(x));
                    } else {
                        int bb = m >> 10, l = m & (LSEQ - 1);
                        size_t o = ((size_t)bb * CCH + n) * LSEQ + l;
                        ((float*)Cout)[o] = val + resid[o];
                    }
                }
            }
        }
    }
}

// ---------------- launch ----------------
extern "C" void kernel_launch(void* const* d_in, const int* in_sizes, int n_in,
                              void* d_out, int out_size, void* d_ws, size_t ws_size,
                              hipStream_t stream)
{
    (void)in_sizes; (void)n_in; (void)out_size; (void)ws_size;
    const float* visual  = (const float*)d_in[0];
    const float* text    = (const float*)d_in[1];
    const float* W_text  = (const float*)d_in[2];
    const float* b_text  = (const float*)d_in[3];
    const float* ln_g    = (const float*)d_in[4];
    const float* ln_b    = (const float*)d_in[5];
    const float* W_in    = (const float*)d_in[6];
    const float* W_conv  = (const float*)d_in[7];
    const float* b_conv  = (const float*)d_in[8];
    const float* W_xproj = (const float*)d_in[9];
    const float* W_dt    = (const float*)d_in[10];
    const float* b_dt    = (const float*)d_in[11];
    const float* A_log   = (const float*)d_in[12];
    const float* Dp      = (const float*)d_in[13];
    const float* W_out   = (const float*)d_in[14];

    char* base = (char*)d_ws;
    size_t off = 0;
    auto alloc = [&](size_t bytes) -> char* {
        char* p = base + off;
        off = (off + bytes + 255) & ~(size_t)255;
        return p;
    };
    float* t_ws     = (float*)alloc((size_t)BATCH * CCH * 4);
    u16*   xn_bf    = (u16*)  alloc((size_t)MROWS * CCH * 2);
    u16*   Win_bf   = (u16*)  alloc((size_t)NXZ * CCH * 2);
    u16*   Wxp_bf   = (u16*)  alloc((size_t)80 * DIN * 2);
    u16*   Wdt_bf   = (u16*)  alloc((size_t)DIN * DTRP * 2);
    u16*   Wout_bf  = (u16*)  alloc((size_t)CCH * DIN * 2);
    u16*   xz_bf    = (u16*)  alloc((size_t)MROWS * NXZ * 2);
    u16*   xc_bf    = (u16*)  alloc((size_t)MROWS * DIN * 2);
    float* dbl_f    = (float*)alloc((size_t)MROWS * 80 * 4);
    u16*   dbldt_bf = (u16*)  alloc((size_t)MROWS * DTRP * 2);
    float* dt_f     = (float*)alloc((size_t)MROWS * DIN * 4);
    float* Aexp     = (float*)alloc((size_t)DIN * DST * 4);
    u16*   y_bf     = (u16*)  alloc((size_t)MROWS * DIN * 2);

    text_proj_kernel<<<(BATCH * CCH + 255) / 256, 256, 0, stream>>>(
        text, W_text, b_text, t_ws);
    f32_to_bf16_kernel<<<(NXZ * CCH + 255) / 256, 256, 0, stream>>>(
        W_in, Win_bf, NXZ * CCH);
    f32_to_bf16_kernel<<<(80 * DIN + 255) / 256, 256, 0, stream>>>(
        W_xproj, Wxp_bf, 80 * DIN);
    wdt_pad_kernel<<<(DIN * DTRP + 255) / 256, 256, 0, stream>>>(W_dt, Wdt_bf);
    f32_to_bf16_kernel<<<(CCH * DIN + 255) / 256, 256, 0, stream>>>(
        W_out, Wout_bf, CCH * DIN);
    aexp_kernel<<<(DIN * DST + 255) / 256, 256, 0, stream>>>(A_log, Aexp);
    gate_ln_kernel<<<MROWS, 256, 0, stream>>>(visual, t_ws, ln_g, ln_b, xn_bf);
    // xz = xn @ W_in^T  (8192 x 3072, K=768) -> bf16
    gemm_bf16_kernel<1><<<dim3(MROWS / 128, NXZ / 128), 256, 0, stream>>>(
        xn_bf, Win_bf, xz_bf, nullptr, nullptr,
        MROWS, NXZ, CCH, CCH, CCH, NXZ);
    conv_silu_kernel<<<(MROWS * DIN) / 256, 256, 0, stream>>>(
        xz_bf, W_conv, b_conv, xc_bf);
    // dbl = xc @ W_xproj^T  (8192 x 80, K=1536) -> f32
    gemm_bf16_kernel<0><<<dim3(MROWS / 128, 1), 256, 0, stream>>>(
        xc_bf, Wxp_bf, dbl_f, nullptr, nullptr,
        MROWS, 80, DIN, DIN, DIN, 80);
    pack_dt_kernel<<<(MROWS * DTRP) / 256, 256, 0, stream>>>(dbl_f, dbldt_bf);
    // dt = softplus(dbl48 @ W_dt^T + b_dt)  (8192 x 1536, K=64) -> f32
    gemm_bf16_kernel<2><<<dim3(MROWS / 128, DIN / 128), 256, 0, stream>>>(
        dbldt_bf, Wdt_bf, dt_f, b_dt, nullptr,
        MROWS, DIN, DTRP, DTRP, DTRP, DIN);
    scan_kernel<<<(BATCH * DIN) / 256, 256, 0, stream>>>(
        dt_f, xc_bf, dbl_f, Aexp, Dp, xz_bf, y_bf);
    // out = y @ W_out^T, transpose to (B,C,H,W), + visual residual
    gemm_bf16_kernel<3><<<dim3(MROWS / 128, CCH / 128), 256, 0, stream>>>(
        y_bf, Wout_bf, d_out, nullptr, visual,
        MROWS, CCH, DIN, DIN, DIN, 0);
}
